// DCNv4PackFlowGuided_30812095381654
// MI455X (gfx1250) — matrix operand order repaired and checked by reading, compiled, not run on hardware
//
#include <hip/hip_runtime.h>

// ---------------------------------------------------------------------------
// DCNv4 flow-guided block for MI455X (gfx1250), wave32 + WMMA f16->f32.
//   GEMM1: [65536 x 224] @ [224 x 608] (value + offset/mask fused, N pad 640)
//   GEMM2: [65536 x 224] @ [224 x 64]  (only first 64 out channels survive)
// Wave tile 32M x 64N, B tile staged in LDS (28KB) shared by 8 waves/block,
// A fragments register-pipelined across K-steps.
// ---------------------------------------------------------------------------

typedef _Float16 v8h  __attribute__((ext_vector_type(8)));
typedef _Float16 v16h __attribute__((ext_vector_type(16)));
typedef float    v8f  __attribute__((ext_vector_type(8)));

#define MTOT   65536      // B*H*W
#define KDIM   224        // C_ENH
#define N1     608
#define NP1    640
#define NOUT   64
#define HH     128
#define WW     128
#define HW     16384
#define NGRP   14
#define CG     16

// workspace layout (all 256B aligned)
#define OFF_INP  ((size_t)0)                       // [MTOT][KDIM] f16 (reused as samp)
#define OFF_Y1   ((size_t)29360128)                // [MTOT][NP1]  f16
#define OFF_WC   ((size_t)113246208)               // [NP1][KDIM]  f16 (K-contiguous)
#define OFF_WO   ((size_t)113532928)               // [NOUT][KDIM] f16
#define OFF_BC   ((size_t)113561600)               // [NP1] f32
#define OFF_BO   ((size_t)113564160)               // [NOUT] f32

// -------------------------------- weight pack ------------------------------
__global__ __launch_bounds__(256) void wpack_kernel(
    const float* __restrict__ value_w, const float* __restrict__ value_b,
    const float* __restrict__ offset_w, const float* __restrict__ offset_b,
    const float* __restrict__ output_w, const float* __restrict__ output_b,
    _Float16* __restrict__ Wc, _Float16* __restrict__ Wo,
    float* __restrict__ biasC, float* __restrict__ biasO) {
  int idx = blockIdx.x * 256 + threadIdx.x;
  if (idx < NP1 * KDIM) {                       // combined GEMM1 weights, transposed
    int k = idx % KDIM, n = idx / KDIM;
    float v = 0.f;
    if (n < 224)      v = value_w[k * 224 + n];
    else if (n < 608) v = offset_w[k * 384 + (n - 224)];
    Wc[(size_t)n * KDIM + k] = (_Float16)v;
    return;
  }
  idx -= NP1 * KDIM;
  if (idx < NOUT * KDIM) {                      // output weights (first 64 cols)
    int k = idx % KDIM, n = idx / KDIM;
    Wo[(size_t)n * KDIM + k] = (_Float16)output_w[k * 224 + n];
    return;
  }
  idx -= NOUT * KDIM;
  if (idx < NP1) {
    biasC[idx] = (idx < 224) ? value_b[idx] : ((idx < 608) ? offset_b[idx - 224] : 0.f);
    return;
  }
  idx -= NP1;
  if (idx < NOUT) biasO[idx] = output_b[idx];
}

// ------------------------- NCHW concat -> [M][K] f16 -----------------------
__global__ __launch_bounds__(256) void pack_kernel(
    const float* __restrict__ x, const float* __restrict__ xw,
    const float* __restrict__ xc, const float* __restrict__ flow,
    _Float16* __restrict__ inpH) {
  size_t idx = (size_t)blockIdx.x * 256 + threadIdx.x;   // exactly MTOT*KDIM threads
  int c = (int)(idx % KDIM);
  size_t m = idx / KDIM;
  int b = (int)(m >> 14);
  int pix = (int)(m & (HW - 1));
  float v;
  if (c < 64)       v = x   [((size_t)(b * 64 + c      )) * HW + pix];
  else if (c < 128) v = xw  [((size_t)(b * 64 + c - 64 )) * HW + pix];
  else if (c < 192) v = xc  [((size_t)(b * 64 + c - 128)) * HW + pix];
  else              v = flow[((size_t)(b * 2 + ((c - 192) & 1))) * HW + pix];
  inpH[idx] = (_Float16)v;
}

// A fragment (16x32 f16): lanes 0-15 rows M0..15 K[0..7]/K[16..23],
// lanes 16-31 same rows K[8..15]/K[24..31]  (ISA 7.12.2 16-bit A layout)
__device__ __forceinline__ v16h load_a_frag(const _Float16* __restrict__ row, int ka) {
  v8h lo = *(const v8h*)(row + ka);
  v8h hi = *(const v8h*)(row + ka + 16);
  return __builtin_shufflevector(lo, hi, 0,1,2,3,4,5,6,7,8,9,10,11,12,13,14,15);
}

__device__ __forceinline__ v16h load_b_lds(const _Float16* p) {
  v8h lo = *(const v8h*)p;
  v8h hi = *(const v8h*)(p + 16);
  return __builtin_shufflevector(lo, hi, 0,1,2,3,4,5,6,7,8,9,10,11,12,13,14,15);
}

#define WMMA_F16(A, B, C) __builtin_amdgcn_wmma_f32_16x16x32_f16(false, (A), false, (B), (short)0, (C), false, false)

// ----------------------- GEMM1: inp @ [Wval|Woff] --------------------------
// block = 8 waves sharing one 64-col N-tile; wave tile = 32M x 64N.
__global__ __launch_bounds__(256) void gemm1_kernel(
    const _Float16* __restrict__ A, const _Float16* __restrict__ Bw,
    const float* __restrict__ bias, _Float16* __restrict__ Y) {
  __shared__ __align__(32) _Float16 Bs[64 * KDIM];   // 28,672 B

  const int tid  = threadIdx.x;
  const int lane = tid & 31;
  const int wv   = tid >> 5;
  const int mblk  = blockIdx.x / 10;     // 256 rows per block
  const int ntile = blockIdx.x % 10;
  const int n0 = ntile << 6;

  { // cooperative stage of the 64x224 f16 B tile: 1792 b128 chunks
    const v8h* src = (const v8h*)(Bw + (size_t)n0 * KDIM);
    v8h* dst = (v8h*)Bs;
#pragma unroll
    for (int i = 0; i < 7; ++i) dst[tid + i * 256] = src[tid + i * 256];
  }
  __syncthreads();

  const int m0   = (mblk << 8) + (wv << 5);   // 32 rows per wave
  const int lrow = lane & 15;
  const int lhi  = lane >> 4;
  const _Float16* arow0 = A + (size_t)(m0 + lrow) * KDIM;
  const _Float16* arow1 = arow0 + (size_t)16 * KDIM;

  v8f acc00 = {}, acc01 = {}, acc02 = {}, acc03 = {};   // rows 0-15
  v8f acc10 = {}, acc11 = {}, acc12 = {}, acc13 = {};   // rows 16-31

  v16h a0 = load_a_frag(arow0, lhi * 8);
  v16h a1 = load_a_frag(arow1, lhi * 8);

#pragma unroll
  for (int k0 = 0; k0 < KDIM; k0 += 32) {
    v16h na0 = a0, na1 = a1;
    if (k0 + 32 < KDIM) {                 // prefetch next K-step's A fragments
      na0 = load_a_frag(arow0, k0 + 32 + lhi * 8);
      na1 = load_a_frag(arow1, k0 + 32 + lhi * 8);
    }
    const _Float16* bp = Bs + (size_t)lrow * KDIM + (k0 + lhi * 8);
    v16h b0 = load_b_lds(bp);
    v16h b1 = load_b_lds(bp + 16 * KDIM);
    v16h b2 = load_b_lds(bp + 32 * KDIM);
    v16h b3 = load_b_lds(bp + 48 * KDIM);
    acc00 = WMMA_F16(a0, b0, acc00);
    acc01 = WMMA_F16(a0, b1, acc01);
    acc02 = WMMA_F16(a0, b2, acc02);
    acc03 = WMMA_F16(a0, b3, acc03);
    acc10 = WMMA_F16(a1, b0, acc10);
    acc11 = WMMA_F16(a1, b1, acc11);
    acc12 = WMMA_F16(a1, b2, acc12);
    acc13 = WMMA_F16(a1, b3, acc13);
    a0 = na0;
    a1 = na1;
  }

  // C layout: VGPR r -> M = r + 8*(lane>=16)
#define G1_STORE(T, ACC, ROFF)                                               \
  {                                                                          \
    int ncol = n0 + (T) * 16 + lrow;                                         \
    float bv = bias[ncol];                                                   \
    _Float16* dst = Y + (size_t)(m0 + (ROFF) + lhi * 8) * NP1 + ncol;        \
    _Pragma("unroll")                                                        \
    for (int r = 0; r < 8; ++r) dst[(size_t)r * NP1] = (_Float16)((ACC)[r] + bv); \
  }
  G1_STORE(0, acc00, 0)  G1_STORE(1, acc01, 0)
  G1_STORE(2, acc02, 0)  G1_STORE(3, acc03, 0)
  G1_STORE(0, acc10, 16) G1_STORE(1, acc11, 16)
  G1_STORE(2, acc12, 16) G1_STORE(3, acc13, 16)
#undef G1_STORE
}

// ------------------------- DCNv4 bilinear sampling -------------------------
// one thread per (pixel m, group g): 9 taps * 4 corners * 16 channels.
__global__ __launch_bounds__(256) void sample_kernel(
    const _Float16* __restrict__ Y1, _Float16* __restrict__ sampH) {
  int tid = blockIdx.x * 256 + threadIdx.x;    // exactly MTOT*NGRP threads
  int g = tid % NGRP;
  int m = tid / NGRP;
  int b = m >> 14;
  int pix = m & (HW - 1);
  int h = pix >> 7;
  int w = pix & (WW - 1);

  const _Float16* om = Y1 + (size_t)m * NP1 + 224 + g * 27;      // dx,dy x9, mask x9
  const _Float16* vbase = Y1 + (size_t)b * HW * NP1 + g * CG;    // value plane

  float acc[CG];
#pragma unroll
  for (int c = 0; c < CG; ++c) acc[c] = 0.f;

#pragma unroll
  for (int k = 0; k < 9; ++k) {
    const int ki = k / 3, kj = k % 3;
    float dx = (float)om[2 * k];
    float dy = (float)om[2 * k + 1];
    float mk = (float)om[18 + k];
    float py = (float)(h - 1 + ki) + dy;
    float px = (float)(w - 1 + kj) + dx;
    float y0 = floorf(py), x0 = floorf(px);
    float wy = py - y0, wx = px - x0;
    int y0i = (int)y0, x0i = (int)x0;
#pragma unroll
    for (int sy = 0; sy < 2; ++sy) {
#pragma unroll
      for (int sx = 0; sx < 2; ++sx) {
        int yy = y0i + sy, xx = x0i + sx;
        if (yy >= 0 && yy < HH && xx >= 0 && xx < WW) {
          float wt = mk * (sy ? wy : 1.f - wy) * (sx ? wx : 1.f - wx);
          v16h v = *(const v16h*)(vbase + (size_t)(yy * WW + xx) * NP1);
#pragma unroll
          for (int c = 0; c < CG; ++c) acc[c] += wt * (float)v[c];
        }
      }
    }
  }

  v16h outv;
#pragma unroll
  for (int c = 0; c < CG; ++c) outv[c] = (_Float16)acc[c];
  *(v16h*)(sampH + (size_t)m * KDIM + g * CG) = outv;
}

// ----------------- GEMM2: samp @ Wout[:, :64] -> NCHW f32 ------------------
__global__ __launch_bounds__(256) void gemm2_kernel(
    const _Float16* __restrict__ A, const _Float16* __restrict__ Bw,
    const float* __restrict__ bias, float* __restrict__ Out) {
  __shared__ __align__(32) _Float16 Bs[64 * KDIM];

  const int tid  = threadIdx.x;
  const int lane = tid & 31;
  const int wv   = tid >> 5;

  { // stage the full 64x224 output-weight tile
    const v8h* src = (const v8h*)Bw;
    v8h* dst = (v8h*)Bs;
#pragma unroll
    for (int i = 0; i < 7; ++i) dst[tid + i * 256] = src[tid + i * 256];
  }
  __syncthreads();

  const int m0   = (blockIdx.x << 8) + (wv << 5);   // 32 rows per wave
  const int lrow = lane & 15;
  const int lhi  = lane >> 4;
  const _Float16* arow0 = A + (size_t)(m0 + lrow) * KDIM;
  const _Float16* arow1 = arow0 + (size_t)16 * KDIM;

  v8f acc00 = {}, acc01 = {}, acc02 = {}, acc03 = {};
  v8f acc10 = {}, acc11 = {}, acc12 = {}, acc13 = {};

  v16h a0 = load_a_frag(arow0, lhi * 8);
  v16h a1 = load_a_frag(arow1, lhi * 8);

#pragma unroll
  for (int k0 = 0; k0 < KDIM; k0 += 32) {
    v16h na0 = a0, na1 = a1;
    if (k0 + 32 < KDIM) {
      na0 = load_a_frag(arow0, k0 + 32 + lhi * 8);
      na1 = load_a_frag(arow1, k0 + 32 + lhi * 8);
    }
    const _Float16* bp = Bs + (size_t)lrow * KDIM + (k0 + lhi * 8);
    v16h b0 = load_b_lds(bp);
    v16h b1 = load_b_lds(bp + 16 * KDIM);
    v16h b2 = load_b_lds(bp + 32 * KDIM);
    v16h b3 = load_b_lds(bp + 48 * KDIM);
    acc00 = WMMA_F16(a0, b0, acc00);
    acc01 = WMMA_F16(a0, b1, acc01);
    acc02 = WMMA_F16(a0, b2, acc02);
    acc03 = WMMA_F16(a0, b3, acc03);
    acc10 = WMMA_F16(a1, b0, acc10);
    acc11 = WMMA_F16(a1, b1, acc11);
    acc12 = WMMA_F16(a1, b2, acc12);
    acc13 = WMMA_F16(a1, b3, acc13);
    a0 = na0;
    a1 = na1;
  }

#define G2_STORE(T, ACC, ROFF)                                               \
  {                                                                          \
    int ncol = (T) * 16 + lrow;                                              \
    float bv = bias[ncol];                                                   \
    _Pragma("unroll")                                                        \
    for (int r = 0; r < 8; ++r) {                                            \
      int m = m0 + (ROFF) + lhi * 8 + r;                                     \
      int b = m >> 14;                                                       \
      int pix = m & (HW - 1);                                                \
      Out[((size_t)(b * NOUT + ncol)) * HW + pix] = (ACC)[r] + bv;           \
    }                                                                        \
  }
  G2_STORE(0, acc00, 0)  G2_STORE(1, acc01, 0)
  G2_STORE(2, acc02, 0)  G2_STORE(3, acc03, 0)
  G2_STORE(0, acc10, 16) G2_STORE(1, acc11, 16)
  G2_STORE(2, acc12, 16) G2_STORE(3, acc13, 16)
#undef G2_STORE
}

// ---------------------------------------------------------------------------
extern "C" void kernel_launch(void* const* d_in, const int* in_sizes, int n_in,
                              void* d_out, int out_size, void* d_ws, size_t ws_size,
                              hipStream_t stream) {
  (void)in_sizes; (void)n_in; (void)out_size; (void)ws_size;
  const float* x        = (const float*)d_in[0];
  const float* xw       = (const float*)d_in[1];
  const float* xc       = (const float*)d_in[2];
  const float* flow     = (const float*)d_in[3];
  const float* value_w  = (const float*)d_in[4];
  const float* value_b  = (const float*)d_in[5];
  const float* offset_w = (const float*)d_in[6];
  const float* offset_b = (const float*)d_in[7];
  const float* output_w = (const float*)d_in[8];
  const float* output_b = (const float*)d_in[9];

  char* ws = (char*)d_ws;
  _Float16* inpH  = (_Float16*)(ws + OFF_INP);   // [MTOT][KDIM], reused as samp
  _Float16* Y1    = (_Float16*)(ws + OFF_Y1);    // [MTOT][NP1]
  _Float16* Wc    = (_Float16*)(ws + OFF_WC);
  _Float16* Wo    = (_Float16*)(ws + OFF_WO);
  float*    biasC = (float*)(ws + OFF_BC);
  float*    biasO = (float*)(ws + OFF_BO);

  wpack_kernel<<<(NP1 * KDIM + NOUT * KDIM + NP1 + NOUT + 255) / 256, 256, 0, stream>>>(
      value_w, value_b, offset_w, offset_b, output_w, output_b, Wc, Wo, biasC, biasO);

  // pack: MTOT*KDIM = 14,680,064 elems = 57344 blocks exactly
  pack_kernel<<<(MTOT * KDIM) / 256, 256, 0, stream>>>(x, xw, xc, flow, inpH);

  // GEMM1: 256 row-blocks * 10 N-tiles = 2560 blocks, 8 waves each
  gemm1_kernel<<<2560, 256, 0, stream>>>(inpH, Wc, biasC, Y1);

  // sampling: MTOT*NGRP = 917504 threads = 3584 blocks exactly
  sample_kernel<<<(MTOT * NGRP) / 256, 256, 0, stream>>>(Y1, inpH /*sampH*/);

  // GEMM2: 256 blocks of 8 waves x 32 rows
  gemm2_kernel<<<256, 256, 0, stream>>>(inpH, Wo, biasO, (float*)d_out);
}
